// BiLSTM_CRF_8701603742195
// MI455X (gfx1250) — matrix operand order repaired and checked by reading, not compiled, not run on hardware
//
#include <hip/hip_runtime.h>
#include <cstddef>
#include <cstdint>

// BiLSTM-CRF for MI455X (gfx1250), wave32 + WMMA f16->f32 + TDM async staging.
//
// Pipeline:
//  0) cvt_kernel: f32->f16 weight conversion (Wih_f/b, Whh_f/b, W_lin)
//  1) input_gemm_kernel: Xg[dir] = emb(tokens) @ Wih^T + bih + bhh  (time-parallel,
//     stored f16 in C-tile-swizzled layout -> each step's slice is 128KB contiguous)
//  2) lstm_scan_kernel: 2 blocks (fwd/bwd), 32 waves each; h ping-pong in LDS,
//     per-step [64,256]@[256,1024] via v_wmma_f32_16x16x32_f16; next step's Xg
//     slice is DMA'd into LDS by the Tensor Data Mover (tensor_load_to_lds)
//     overlapped with the WMMA k-loop, fenced with s_wait_tensorcnt.
//  3) emit_crf_kernel: h_cat @ W_lin^T + b_lin, then +transition broadcast -> out

typedef __attribute__((ext_vector_type(16))) _Float16 v16h;
typedef __attribute__((ext_vector_type(8)))  _Float16 v8h;
typedef __attribute__((ext_vector_type(8)))  float    v8f;
typedef __attribute__((ext_vector_type(4)))  unsigned int u32x4;
typedef __attribute__((ext_vector_type(8)))  int i32x8;
typedef __attribute__((ext_vector_type(4)))  int i32x4;

union V16U { v16h v; v8h h[2]; };

#define LQ   512
#define BQ   64
#define EMBD 128
#define HIDD 256
#define G4   1024
#define NTAG 32
#define FEAT 512   // 2*HIDD

#define XG_STEP_BYTES  (BQ * G4 * 2)       // 131072 B per time step
#define XG_STEP_QWORDS (XG_STEP_BYTES / 8) // 16384 x 8B

static __device__ __forceinline__ v8f wmma16(v16h a, v16h b, v8f c) {
  // (neg_a, A, neg_b, B, c_mod, C, reuse_a, reuse_b)
  return __builtin_amdgcn_wmma_f32_16x16x32_f16(false, a, false, b, (short)0, c,
                                                false, false);
}

// ---- Tensor Data Mover: 1-D contiguous copy global -> LDS (D# per ISA ch.8) ----
static __device__ __forceinline__ void tdm_load_1d(uint32_t lds_off,
                                                   const void* gsrc,
                                                   uint32_t n_qwords) {
  uint64_t ga = (uint64_t)(uintptr_t)gsrc;
  u32x4 g0;
  g0[0] = 1u;                                                 // count=1 (valid D#)
  g0[1] = lds_off;                                            // lds_addr (bytes)
  g0[2] = (uint32_t)ga;                                       // global_addr[31:0]
  g0[3] = (uint32_t)((ga >> 32) & 0x01FFFFFFu) | (2u << 30);  // addr[56:32] | type=2
  i32x8 g1;
  g1[0] = (int)(3u << 16);                           // data_size = 3 (8 bytes/elem)
  g1[1] = (int)((n_qwords & 0xFFFFu) << 16);         // tensor_dim0[15:0]  @bits 63:48
  g1[2] = (int)((n_qwords >> 16) | (1u << 16));      // tensor_dim0[31:16], tensor_dim1=1
  g1[3] = (int)(n_qwords << 16);                     // tile_dim0 @bits 127:112
  g1[4] = 0;                                         // tile_dim1=0, tile_dim2=0
  g1[5] = (int)n_qwords;                             // tensor_dim0_stride[31:0]
  g1[6] = 0;
  g1[7] = 0;
  i32x4 gz  = {0, 0, 0, 0};                          // groups 2/3 unused (<=2D)
  i32x8 gz8 = {0, 0, 0, 0, 0, 0, 0, 0};              // extra operand (unused here)
  __builtin_amdgcn_tensor_load_to_lds(g0, g1, gz, gz, gz8, 0);
}

// B-matrix tile (32x16, K x N) for D = A*B + C, where B = W^T and W is
// row-major [N][Kpitch]: B[k][n] = W[nrow0+n][k].
// ISA layout: lanes 0-15 hold col N=lane, K {0..7, 16..23}; lanes 16-31 K {8..15, 24..31}.
static __device__ __forceinline__ v16h load_b_tile(const _Float16* W, int pitch,
                                                   int nrow0, int kbase, int lane) {
  int n    = lane & 15;
  int kofs = (lane < 16) ? 0 : 8;
  const _Float16* q = W + (size_t)(nrow0 + n) * pitch + (kbase + kofs);
  V16U u;
  u.h[0] = *(const v8h*)(q);
  u.h[1] = *(const v8h*)(q + 16);
  return u.v;
}

// A-matrix tile (16x32, M x K) from row-major [M][pitch] f16.
// ISA layout: lanes 0-15 hold row M=lane, K {0..7, 16..23};
//             lanes 16-31 hold row M=lane-16, K {8..15, 24..31}.
static __device__ __forceinline__ v16h load_a_tile(const _Float16* A, int pitch,
                                                   int row0, int kbase, int lane) {
  int m    = lane & 15;
  int kofs = (lane < 16) ? 0 : 8;
  const _Float16* p = A + (size_t)(row0 + m) * pitch + (kbase + kofs);
  V16U u;
  u.h[0] = *(const v8h*)(p);
  u.h[1] = *(const v8h*)(p + 16);
  return u.v;
}

// ------------------------- weight conversion -------------------------
__global__ void cvt_kernel(const float* __restrict__ in,
                           _Float16* __restrict__ out, int n) {
  int i = blockIdx.x * blockDim.x + threadIdx.x;
  if (i < n) out[i] = (_Float16)in[i];
}

// ------------------------- input projection --------------------------
// Xg swizzled layout (f16): [mtile(0..2047)][col(0..1023)][m(0..15)]
// row r = l*64 + b  (time-major rows), mtile = r/16.
__global__ __launch_bounds__(256) void input_gemm_kernel(
    const int* __restrict__ sents, const float* __restrict__ emb,
    const _Float16* __restrict__ Wih, const float* __restrict__ bih,
    const float* __restrict__ bhh, _Float16* __restrict__ XgSw)
{
  const int lane  = threadIdx.x & 31;
  const int wave  = threadIdx.x >> 5;
  const int mtile = blockIdx.x;                 // 0..2047
  const int nlane = lane & 15;
  const int mofs  = (lane < 16) ? 0 : 8;
  const int akofs = (lane < 16) ? 0 : 8;

  // this lane's A row -> embedding row
  int r = mtile * 16 + nlane;
  int l = r >> 6, b = r & 63;
  int tok = sents[b * LQ + l];
  const float* erow = emb + (size_t)tok * EMBD;

  v8f acc[8];
  #pragma unroll
  for (int j = 0; j < 8; ++j) {
    int n = (wave * 8 + j) * 16 + nlane;
    float bias = bih[n] + bhh[n];
    #pragma unroll
    for (int v = 0; v < 8; ++v) acc[j][v] = bias;
  }

  #pragma unroll
  for (int kt = 0; kt < 4; ++kt) {
    const float* p = erow + kt * 32 + akofs;
    v16h A;
    #pragma unroll
    for (int i = 0; i < 8; ++i) {
      A[i]     = (_Float16)p[i];
      A[i + 8] = (_Float16)p[i + 16];
    }
    #pragma unroll
    for (int j = 0; j < 8; ++j) {
      int n0 = (wave * 8 + j) * 16;
      v16h B = load_b_tile(Wih, EMBD, n0, kt * 32, lane);
      acc[j] = wmma16(A, B, acc[j]);
    }
  }

  #pragma unroll
  for (int j = 0; j < 8; ++j) {
    int col = (wave * 8 + j) * 16 + nlane;
    v8h hv;
    #pragma unroll
    for (int v = 0; v < 8; ++v) hv[v] = (_Float16)acc[j][v];
    *(v8h*)(XgSw + (((size_t)mtile * G4 + col) << 4) + mofs) = hv;
  }
}

// ------------------------- recurrent scan ----------------------------
// Dynamic LDS: hb[2][64][256] f16 (64KB ping-pong of h) + xs[4][1024][16] f16
// (128KB Xg stage filled by the Tensor Data Mover) = 192KB of the 320KB WGP LDS.
__global__ __launch_bounds__(1024) void lstm_scan_kernel(
    const _Float16* __restrict__ XgF, const _Float16* __restrict__ XgB,
    const _Float16* __restrict__ WhhF, const _Float16* __restrict__ WhhB,
    _Float16* __restrict__ hcat)
{
  extern __shared__ _Float16 smem[];
  _Float16* hb = smem;                       // [2][BQ][HIDD]
  _Float16* xs = smem + 2 * BQ * HIDD;       // [4][G4][16] staged step slice

  const int dir = blockIdx.x;                // 0 = forward, 1 = backward
  const _Float16* __restrict__ Xg  = dir ? XgB  : XgF;
  const _Float16* __restrict__ Whh = dir ? WhhB : WhhF;
  const int hofs = dir ? HIDD : 0;

  const int tid   = threadIdx.x;
  const int lane  = tid & 31;
  const int wave  = tid >> 5;                // 0..31
  const int hs    = wave & 15;               // hidden-slice tile (16 units)
  const int mh    = wave >> 4;               // batch half (m-tiles mh*2, mh*2+1)
  const int nlane = lane & 15;
  const int mofs  = (lane < 16) ? 0 : 8;

  const uint32_t xs_base = (uint32_t)(uintptr_t)(void*)xs;

  // kick off the TDM for step 0 while zeroing h, then fence + publish
  if (wave == 0) {
    int t0 = dir ? (LQ - 1) : 0;
    tdm_load_1d(xs_base, (const char*)Xg + (size_t)t0 * XG_STEP_BYTES,
                XG_STEP_QWORDS);
  }
  for (int i = tid; i < BQ * HIDD; i += 1024) hb[i] = (_Float16)0.0f;
  if (wave == 0) __builtin_amdgcn_s_wait_tensorcnt(0);
  __syncthreads();

  v8f cst[2];
  #pragma unroll
  for (int m = 0; m < 2; ++m)
    #pragma unroll
    for (int v = 0; v < 8; ++v) cst[m][v] = 0.0f;

  int p = 0;
  for (int s = 0; s < LQ; ++s) {
    const int t = dir ? (LQ - 1 - s) : s;

    // accumulator init from staged gates (already include biases)
    v8f acc[4][2];
    #pragma unroll
    for (int g = 0; g < 4; ++g) {
      #pragma unroll
      for (int m = 0; m < 2; ++m) {
        int mtl = mh * 2 + m;
        int col = g * HIDD + hs * 16 + nlane;
        v8h xv = *(const v8h*)(xs + (((mtl * G4 + col) << 4) + mofs));
        #pragma unroll
        for (int v = 0; v < 8; ++v) acc[g][m][v] = (float)xv[v];
      }
    }

    __syncthreads();   // whole block done reading the stage buffer

    // DMA next step's 128KB gate slice, overlapped with the WMMA k-loop below
    if (wave == 0 && s + 1 < LQ) {
      int tn = dir ? (LQ - 2 - s) : (s + 1);
      tdm_load_1d(xs_base, (const char*)Xg + (size_t)tn * XG_STEP_BYTES,
                  XG_STEP_QWORDS);
    }

    // gates += h_prev @ Whh^T  (A from LDS, B streamed from L2-resident Whh)
    #pragma unroll
    for (int kt = 0; kt < 8; ++kt) {
      v16h A0 = load_a_tile(hb + (size_t)p * BQ * HIDD, HIDD,
                            (mh * 2 + 0) * 16, kt * 32, lane);
      v16h A1 = load_a_tile(hb + (size_t)p * BQ * HIDD, HIDD,
                            (mh * 2 + 1) * 16, kt * 32, lane);
      #pragma unroll
      for (int g = 0; g < 4; ++g) {
        v16h B = load_b_tile(Whh, HIDD, g * HIDD + hs * 16, kt * 32, lane);
        acc[g][0] = wmma16(A0, B, acc[g][0]);
        acc[g][1] = wmma16(A1, B, acc[g][1]);
      }
    }

    // LSTM cell (f32), write h -> other LDS buffer + global h_cat
    #pragma unroll
    for (int m = 0; m < 2; ++m) {
      int mt   = mh * 2 + m;
      int hcol = hs * 16 + nlane;
      _Float16* lb = hb + ((size_t)(p ^ 1) * BQ + mt * 16 + mofs) * HIDD + hcol;
      _Float16* gb = hcat + ((size_t)t * BQ + mt * 16 + mofs) * FEAT + hofs + hcol;
      #pragma unroll
      for (int v = 0; v < 8; ++v) {
        float ig = 1.0f / (1.0f + __expf(-acc[0][m][v]));
        float fg = 1.0f / (1.0f + __expf(-acc[1][m][v]));
        float gg = tanhf(acc[2][m][v]);
        float og = 1.0f / (1.0f + __expf(-acc[3][m][v]));
        float c  = fg * cst[m][v] + ig * gg;
        float h  = og * tanhf(c);
        cst[m][v] = c;
        _Float16 hh = (_Float16)h;
        lb[(size_t)v * HIDD] = hh;
        gb[(size_t)v * FEAT] = hh;
      }
    }

    if (wave == 0) __builtin_amdgcn_s_wait_tensorcnt(0);  // next slice landed
    __syncthreads();                                      // publish h + stage
    p ^= 1;
  }
}

// ------------------------- emission + CRF broadcast ------------------
__global__ __launch_bounds__(256) void emit_crf_kernel(
    const _Float16* __restrict__ hcat, const _Float16* __restrict__ Wlin,
    const float* __restrict__ blin, const float* __restrict__ trans,
    float* __restrict__ out)
{
  const int lane  = threadIdx.x & 31;
  const int wave  = threadIdx.x >> 5;
  const int mtile = blockIdx.x * 8 + wave;   // 0..2047
  const int rbase = mtile * 16;
  const int nlane = lane & 15;
  const int mofs  = (lane < 16) ? 0 : 8;

  v8f acc[2];
  #pragma unroll
  for (int nt = 0; nt < 2; ++nt) {
    float bias = blin[nt * 16 + nlane];
    #pragma unroll
    for (int v = 0; v < 8; ++v) acc[nt][v] = bias;
  }

  #pragma unroll 4
  for (int kt = 0; kt < 16; ++kt) {
    v16h A = load_a_tile(hcat, FEAT, rbase, kt * 32, lane);
    #pragma unroll
    for (int nt = 0; nt < 2; ++nt) {
      v16h B = load_b_tile(Wlin, FEAT, nt * 16, kt * 32, lane);
      acc[nt] = wmma16(A, B, acc[nt]);
    }
  }

  #pragma unroll
  for (int nt = 0; nt < 2; ++nt) {
    int j = nt * 16 + nlane;
    float tj[32];
    #pragma unroll
    for (int i = 0; i < 32; ++i) tj[i] = trans[i * 32 + j];
    #pragma unroll
    for (int v = 0; v < 8; ++v) {
      int r = rbase + mofs + v;
      int l = r >> 6, b = r & 63;
      float e = acc[nt][v];
      float* ob = out + (((size_t)b * LQ + l) * NTAG) * NTAG + j;
      #pragma unroll
      for (int i = 0; i < 32; ++i) ob[(size_t)i * NTAG] = e + tj[i];
    }
  }
}

// ------------------------- host driver -------------------------------
extern "C" void kernel_launch(void* const* d_in, const int* in_sizes, int n_in,
                              void* d_out, int out_size, void* d_ws, size_t ws_size,
                              hipStream_t stream) {
  (void)in_sizes; (void)n_in; (void)out_size; (void)ws_size;

  const int*   sents = (const int*)d_in[0];
  // d_in[1] = lengths (unused by reference)
  const float* emb   = (const float*)d_in[2];
  const float* WihF  = (const float*)d_in[3];
  const float* WhhF  = (const float*)d_in[4];
  const float* bihF  = (const float*)d_in[5];
  const float* bhhF  = (const float*)d_in[6];
  const float* WihB  = (const float*)d_in[7];
  const float* WhhB  = (const float*)d_in[8];
  const float* bihB  = (const float*)d_in[9];
  const float* bhhB  = (const float*)d_in[10];
  const float* Wlin  = (const float*)d_in[11];
  const float* blin  = (const float*)d_in[12];
  const float* trans = (const float*)d_in[13];
  float* out = (float*)d_out;

  char* ws = (char*)d_ws;
  size_t off = 0;
  auto take = [&](size_t bytes) -> void* {
    size_t a = (off + 255) & ~(size_t)255;
    off = a + bytes;
    return (void*)(ws + a);
  };

  _Float16* WihF16f = (_Float16*)take((size_t)G4 * EMBD * 2);
  _Float16* WhhF16f = (_Float16*)take((size_t)G4 * HIDD * 2);
  _Float16* WihF16b = (_Float16*)take((size_t)G4 * EMBD * 2);
  _Float16* WhhF16b = (_Float16*)take((size_t)G4 * HIDD * 2);
  _Float16* WlinF16 = (_Float16*)take((size_t)NTAG * FEAT * 2);
  _Float16* XgF     = (_Float16*)take((size_t)LQ * BQ * G4 * 2);   // 64 MB
  _Float16* XgB     = (_Float16*)take((size_t)LQ * BQ * G4 * 2);   // 64 MB
  _Float16* hcat    = (_Float16*)take((size_t)LQ * BQ * FEAT * 2); // 32 MB

  // 0) weight conversion
  cvt_kernel<<<(G4 * EMBD + 255) / 256, 256, 0, stream>>>(WihF, WihF16f, G4 * EMBD);
  cvt_kernel<<<(G4 * HIDD + 255) / 256, 256, 0, stream>>>(WhhF, WhhF16f, G4 * HIDD);
  cvt_kernel<<<(G4 * EMBD + 255) / 256, 256, 0, stream>>>(WihB, WihF16b, G4 * EMBD);
  cvt_kernel<<<(G4 * HIDD + 255) / 256, 256, 0, stream>>>(WhhB, WhhF16b, G4 * HIDD);
  cvt_kernel<<<(NTAG * FEAT + 255) / 256, 256, 0, stream>>>(Wlin, WlinF16, NTAG * FEAT);

  // 1) time-parallel input projection (embedding gather fused)
  input_gemm_kernel<<<2048, 256, 0, stream>>>(sents, emb, WihF16f, bihF, bhhF, XgF);
  input_gemm_kernel<<<2048, 256, 0, stream>>>(sents, emb, WihF16b, bihB, bhhB, XgB);

  // 2) bidirectional recurrent scan (block 0 = fwd, block 1 = bwd), 192KB LDS
  lstm_scan_kernel<<<2, 1024, 196608, stream>>>(XgF, XgB, WhhF16f, WhhF16b, hcat);

  // 3) emission GEMM + CRF transition broadcast
  emit_crf_kernel<<<256, 256, 0, stream>>>(hcat, WlinF16, blin, trans, out);
}
